// Framework_31379031065122
// MI455X (gfx1250) — matrix operand order repaired and checked by reading, not compiled, not run on hardware
//
#include <hip/hip_runtime.h>

// ---------------------------------------------------------------------------
// Types
// ---------------------------------------------------------------------------
typedef __attribute__((ext_vector_type(4)))  unsigned int u32x4;
typedef __attribute__((ext_vector_type(16))) __bf16      v16bf;
typedef __attribute__((ext_vector_type(8)))  float       v8f;

union Frag { v16bf v; u32x4 q[2]; };

__device__ __forceinline__ unsigned short f2bf(float x) {
  unsigned u = __builtin_bit_cast(unsigned, x);
  unsigned r = 0x7FFFu + ((u >> 16) & 1u);      // round-to-nearest-even
  return (unsigned short)((u + r) >> 16);
}

// ---------------------------------------------------------------------------
// Converters.
//   cvt_nhwc : f32 NCHW activations -> bf16 NHWC (ci innermost, contiguous)
//   cvt_w    : f32 [co][ci][r] weights -> bf16 [co][r][ci]  (K = r*512 + ci)
// ---------------------------------------------------------------------------
__global__ __launch_bounds__(256) void cvt_nhwc(const float* __restrict__ s,
                                                unsigned short* __restrict__ d,
                                                int HW, int total) {
  int idx = blockIdx.x * 256 + threadIdx.x;     // NHWC order (coalesced writes)
  if (idx >= total) return;
  int ci   = idx & 511;
  int rest = idx >> 9;
  int hw   = rest % HW;
  int n    = rest / HW;
  d[idx] = f2bf(s[((size_t)(n * 512 + ci)) * HW + hw]);
}

__global__ __launch_bounds__(256) void cvt_w(const float* __restrict__ s,
                                             unsigned short* __restrict__ d,
                                             int KHW, int total) {
  int idx = blockIdx.x * 256 + threadIdx.x;     // [co][r][ci] order
  if (idx >= total) return;
  int ci   = idx & 511;
  int rest = idx >> 9;
  int r    = rest % KHW;
  int co   = rest / KHW;
  d[idx] = f2bf(s[((size_t)(co * 512 + ci)) * KHW + r]);
}

// ---------------------------------------------------------------------------
// Implicit-GEMM conv via WMMA (bf16 in, f32 accumulate).
// M = Cout = 512, N = NB*OH*OW, K = KH*KW*512 with k = r*512 + ci.
// Block: 256 threads = 8 wave32s. Block tile 128(M) x 64(N) x 32(K);
// each wave owns a 32x32 patch -> 4 x v_wmma_f32_16x16x32_bf16 per K-step.
// Each 32-wide K chunk sits inside one kernel tap r (512 % 32 == 0), so the
// B gather is a single bounds-checked b128 load per thread from NHWC input.
// ---------------------------------------------------------------------------
struct ConvCfg {
  const unsigned short* X;   // bf16 activations, NHWC [img][IH][IW][512]
  const unsigned short* Wt;  // bf16 weights, [Cout][KH*KW][512] contiguous
  float* Y;                  // f32 output, NCHW (Cout = 512)
  int N, K;
  int OH, OW, IH, IW, KW;
  int padH, padW, dilH, strideW;   // strideH == 1, dilW == 1 for all convs here
  int relu;
};

__global__ __launch_bounds__(256) void conv_gemm_wmma(ConvCfg c) {
  // Row stride 48 halves (96B): b128 fragment reads stay 16B-aligned, banks staggered.
  __shared__ alignas(64) unsigned short As[128][48];
  __shared__ alignas(64) unsigned short Bs[64][48];

  const int tid = threadIdx.x;
  const int mo  = blockIdx.y * 128;        // Cout tile
  const int no  = blockIdx.x * 64;         // output-position tile
  const int OHW = c.OH * c.OW;

  const int lane = tid & 31;
  const int wave = tid >> 5;
  const int msub = (wave >> 1) * 32;       // 0,32,64,96
  const int nsub = (wave & 1) * 32;        // 0,32
  const int lm = lane & 15, lh = lane >> 4;

  // ---- A-load coords: 128x32 halves, 2 x b128 per thread
  const int arow = tid >> 1;
  const int aseg = (tid & 1) * 16;

  // ---- B-load coords: 64x32 halves, 1 x b128 per thread (NHWC contiguous ci)
  const int bn_row = tid >> 2;             // 0..63: N within tile
  const int ciseg  = (tid & 3) * 8;        // ci chunk within 32-wide K step
  const int ngl    = no + bn_row;
  const int img    = ngl / OHW;
  const int hw     = ngl - img * OHW;
  const int oh     = hw / c.OW;
  const int ow     = hw - oh * c.OW;
  const int ih0    = oh - c.padH;                 // strideH == 1
  const int iw0    = ow * c.strideW - c.padW;
  const size_t imgBase = (size_t)img * c.IH * c.IW * 512;

  v8f acc00 = {}, acc01 = {}, acc10 = {}, acc11 = {};

  for (int k0 = 0; k0 < c.K; k0 += 32) {
    // ---- A tile (weights): coalesced, streams with prefetch
    {
      const unsigned short* src = c.Wt + (size_t)(mo + arow) * c.K + (k0 + aseg);
      *(u32x4*)&As[arow][aseg]     = *(const u32x4*)src;
      *(u32x4*)&As[arow][aseg + 8] = *(const u32x4*)(src + 8);
      if (k0 + 32 < c.K)
        __builtin_prefetch(src + 32, 0, 1);
    }
    // ---- B tile: one bounds-checked vector load; r/ci via shifts (Cin==512)
    {
      const int r   = k0 >> 9;             // kernel tap index (uniform)
      const int cib = k0 & 511;
      const int kh  = r / c.KW;            // tiny ints, once per K-step
      const int kw  = r - kh * c.KW;
      const int ih  = ih0 + kh * c.dilH;
      const int iw  = iw0 + kw;
      u32x4 bv = {};
      if ((unsigned)ih < (unsigned)c.IH && (unsigned)iw < (unsigned)c.IW)
        bv = *(const u32x4*)(c.X + imgBase + ((size_t)(ih * c.IW + iw)) * 512 +
                             (cib + ciseg));
      *(u32x4*)&Bs[bn_row][ciseg] = bv;
    }
    __syncthreads();

    // ---- fragments per documented CDNA5 wave32 layouts
    Frag a0, a1, b0, b1;
    a0.q[0] = *(const u32x4*)&As[msub + lm][lh * 8];
    a0.q[1] = *(const u32x4*)&As[msub + lm][lh * 8 + 16];
    a1.q[0] = *(const u32x4*)&As[msub + 16 + lm][lh * 8];
    a1.q[1] = *(const u32x4*)&As[msub + 16 + lm][lh * 8 + 16];
    b0.q[0] = *(const u32x4*)&Bs[nsub + lm][lh * 8];
    b0.q[1] = *(const u32x4*)&Bs[nsub + lm][lh * 8 + 16];
    b1.q[0] = *(const u32x4*)&Bs[nsub + 16 + lm][lh * 8];
    b1.q[1] = *(const u32x4*)&Bs[nsub + 16 + lm][lh * 8 + 16];

    acc00 = __builtin_amdgcn_wmma_f32_16x16x32_bf16(false, a0.v, false, b0.v,
                                                    (short)0, acc00, false, false);
    acc01 = __builtin_amdgcn_wmma_f32_16x16x32_bf16(false, a0.v, false, b1.v,
                                                    (short)0, acc01, false, false);
    acc10 = __builtin_amdgcn_wmma_f32_16x16x32_bf16(false, a1.v, false, b0.v,
                                                    (short)0, acc10, false, false);
    acc11 = __builtin_amdgcn_wmma_f32_16x16x32_bf16(false, a1.v, false, b1.v,
                                                    (short)0, acc11, false, false);
    __syncthreads();
  }

  // ---- epilogue: scatter to NCHW (Cout = 512)
  const int nA = no + nsub + lm;
  const int nB = nA + 16;
  const int mb0 = mo + msub + lh * 8;
  const int mb1 = mb0 + 16;
  const int iA = nA / OHW, hwA = nA - iA * OHW;
  const int iB = nB / OHW, hwB = nB - iB * OHW;
  float* yA = c.Y + (size_t)iA * 512 * OHW + hwA;
  float* yB = c.Y + (size_t)iB * 512 * OHW + hwB;
#pragma unroll
  for (int r = 0; r < 8; ++r) {
    float v00 = acc00[r], v01 = acc01[r], v10 = acc10[r], v11 = acc11[r];
    if (c.relu) {
      v00 = fmaxf(v00, 0.f); v01 = fmaxf(v01, 0.f);
      v10 = fmaxf(v10, 0.f); v11 = fmaxf(v11, 0.f);
    }
    yA[(size_t)(mb0 + r) * OHW] = v00;
    yB[(size_t)(mb0 + r) * OHW] = v01;
    yA[(size_t)(mb1 + r) * OHW] = v10;
    yB[(size_t)(mb1 + r) * OHW] = v11;
  }
}

// ---------------------------------------------------------------------------
// BatchNorm batch statistics: one block per channel, reduce over NB*S elems.
// ---------------------------------------------------------------------------
__global__ __launch_bounds__(256) void bn_stats(const float* __restrict__ Y,
                                                int C, int S, int NB,
                                                float* __restrict__ mean,
                                                float* __restrict__ var) {
  __shared__ float s1[256], s2[256];
  const int ch = blockIdx.x;
  const int tot = NB * S;
  float a = 0.f, b = 0.f;
  for (int i = threadIdx.x; i < tot; i += 256) {
    int n = i / S, s = i - n * S;
    float v = Y[((size_t)(n * C + ch)) * S + s];
    a += v; b += v * v;
  }
  s1[threadIdx.x] = a; s2[threadIdx.x] = b;
  __syncthreads();
  for (int st = 128; st > 0; st >>= 1) {
    if (threadIdx.x < st) {
      s1[threadIdx.x] += s1[threadIdx.x + st];
      s2[threadIdx.x] += s2[threadIdx.x + st];
    }
    __syncthreads();
  }
  if (threadIdx.x == 0) {
    float m = s1[0] / tot;
    mean[ch] = m;
    var[ch]  = s2[0] / tot - m * m;     // biased variance (training mode)
  }
}

// ---------------------------------------------------------------------------
// Fused BN (affine) -> optional MaxPool(2,1) over time -> ReLU.
// Xb (bf16, NHWC) feeds the next conv; Xf (f32, NCHW) is the feat_a output.
// ---------------------------------------------------------------------------
__global__ __launch_bounds__(256) void bn_pool_relu(
    const float* __restrict__ Y, const float* __restrict__ mean,
    const float* __restrict__ var, const float* __restrict__ gam,
    const float* __restrict__ bet, unsigned short* __restrict__ Xb,
    float* __restrict__ Xf, int C, int Tin, int F, int poolT, int total) {
  int idx = blockIdx.x * 256 + threadIdx.x;     // NCHW order of pooled tensor
  if (idx >= total) return;
  const int To = Tin / poolT;
  int f  = idx % F;
  int t  = (idx / F) % To;
  int ch = (idx / (F * To)) % C;
  int nb = idx / (F * To * C);
  float sc = rsqrtf(var[ch] + 1e-5f) * gam[ch];
  float sh = bet[ch] - mean[ch] * sc;
  size_t base = ((size_t)(nb * C + ch) * Tin + t * poolT) * F + f;
  float v = Y[base] * sc + sh;                  // BN first (matches reference)
  if (poolT == 2) v = fmaxf(v, Y[base + F] * sc + sh);
  v = fmaxf(v, 0.f);
  if (Xb) Xb[(((size_t)nb * To + t) * F + f) * C + ch] = f2bf(v);   // NHWC
  if (Xf) Xf[idx] = v;                                              // NCHW
}

// ---------------------------------------------------------------------------
// Global max over spatial: one block per (image, channel).
// ---------------------------------------------------------------------------
__global__ __launch_bounds__(64) void max_reduce(const float* __restrict__ X,
                                                 float* __restrict__ out, int S) {
  __shared__ float sm[64];
  const float* p = X + (size_t)blockIdx.x * S;
  float m = -3.402823e38f;
  for (int i = threadIdx.x; i < S; i += 64) m = fmaxf(m, p[i]);
  sm[threadIdx.x] = m;
  __syncthreads();
  for (int st = 32; st > 0; st >>= 1) {
    if (threadIdx.x < st) sm[threadIdx.x] = fmaxf(sm[threadIdx.x], sm[threadIdx.x + st]);
    __syncthreads();
  }
  if (threadIdx.x == 0) out[blockIdx.x] = sm[0];
}

// ---------------------------------------------------------------------------
// Fused discriminator: both "common" and "differ" heads share the ea.W part.
// One block per (b, frame); thread j computes hidden unit j (128 hidden).
// ---------------------------------------------------------------------------
__global__ __launch_bounds__(128) void discrim_kernel(
    const float* __restrict__ ea, const float* __restrict__ ev,
    const float* __restrict__ w1, const float* __restrict__ b1v,
    const float* __restrict__ w2, const float* __restrict__ b2v,
    const int* __restrict__ randidx, const int* __restrict__ perm,
    float* __restrict__ common, float* __restrict__ differ) {
  __shared__ float H1[128], H2[128];
  const int bf = blockIdx.x;            // 0..127 = b*4 + fr
  const int b  = bf >> 2, fr = bf & 3;
  const int j  = threadIdx.x;           // hidden unit
  const float* pa = ea + (size_t)b * 512;
  const float* pv = ev + (size_t)bf * 512;
  const int sidx = randidx[b * 4 + fr] * 4 + perm[b * 4 + fr];
  const float* ps = ev + (size_t)sidx * 512;
  const float* w  = w1 + (size_t)j * 1024;
  float h1 = b1v[j], h2 = h1;
  for (int ci = 0; ci < 512; ++ci) { float t = w[ci] * pa[ci]; h1 += t; h2 += t; }
  for (int ci = 0; ci < 512; ++ci) {
    float wc = w[512 + ci];
    h1 += wc * pv[ci];
    h2 += wc * ps[ci];
  }
  H1[j] = fmaxf(h1, 0.f);
  H2[j] = fmaxf(h2, 0.f);
  __syncthreads();
  if (j < 2) {
    float o1 = b2v[j], o2 = b2v[j];
    const float* wr = w2 + j * 128;
    for (int t = 0; t < 128; ++t) { o1 += wr[t] * H1[t]; o2 += wr[t] * H2[t]; }
    common[bf * 2 + j] = o1;
    differ[bf * 2 + j] = o2;
  }
}

// ---------------------------------------------------------------------------
// Host orchestration
// ---------------------------------------------------------------------------
static inline void launch_conv(const ConvCfg& cfg, hipStream_t stream) {
  dim3 grid(cfg.N / 64, 512 / 128);
  hipLaunchKernelGGL(conv_gemm_wmma, grid, dim3(256), 0, stream, cfg);
}

extern "C" void kernel_launch(void* const* d_in, const int* in_sizes, int n_in,
                              void* d_out, int out_size, void* d_ws, size_t ws_size,
                              hipStream_t stream) {
  (void)in_sizes; (void)n_in; (void)out_size; (void)ws_size;
  const float* audio  = (const float*)d_in[0];   // [32,512,64,4]
  const float* visual = (const float*)d_in[1];   // [128,512,14,14]
  const float* w_tc1  = (const float*)d_in[2];   // [512,512,3,1]
  const float* g1     = (const float*)d_in[3];
  const float* b1     = (const float*)d_in[4];
  const float* w_tc2  = (const float*)d_in[5];   // [512,512,1,2]
  const float* g2     = (const float*)d_in[6];
  const float* b2     = (const float*)d_in[7];
  const float* w_tc3  = (const float*)d_in[8];   // [512,512,3,1]
  const float* g3     = (const float*)d_in[9];
  const float* b3     = (const float*)d_in[10];
  const float* w_spa  = (const float*)d_in[11];  // [512,512,3,3]
  const float* w_d1   = (const float*)d_in[12];  // [128,1024]
  const float* b_d1   = (const float*)d_in[13];
  const float* w_d2   = (const float*)d_in[14];  // [2,128]
  const float* b_d2   = (const float*)d_in[15];
  const int* randidx  = (const int*)d_in[16];
  const int* perm     = (const int*)d_in[17];

  float* out    = (float*)d_out;
  float* common = out;                       // 256
  float* differ = out + 256;                 // 256
  float* feat_a = out + 512;                 // 32*512*16*2 = 524288
  float* feat_v = out + 512 + 524288;        // 128*512*196 = 12845056

  char* wsb = (char*)d_ws;
  size_t off = 0;
  auto take = [&](size_t bytes) -> void* {
    void* p = wsb + off;
    off += (bytes + 255) & ~(size_t)255;
    return p;
  };
  unsigned short* wb1  = (unsigned short*)take((size_t)786432 * 2);
  unsigned short* wb2  = (unsigned short*)take((size_t)524288 * 2);
  unsigned short* wb3  = (unsigned short*)take((size_t)786432 * 2);
  unsigned short* wbs  = (unsigned short*)take((size_t)2359296 * 2);
  unsigned short* abf  = (unsigned short*)take((size_t)4194304 * 2);
  unsigned short* vbf  = (unsigned short*)take((size_t)12845056 * 2);
  float* y1            = (float*)take((size_t)4194304 * 4);
  unsigned short* x2   = (unsigned short*)take((size_t)2097152 * 2);
  float* y2            = (float*)take((size_t)1048576 * 4);
  unsigned short* x3   = (unsigned short*)take((size_t)1048576 * 2);
  float* y3            = (float*)take((size_t)1048576 * 4);
  float* mean1 = (float*)take(512 * 4); float* var1 = (float*)take(512 * 4);
  float* mean2 = (float*)take(512 * 4); float* var2 = (float*)take(512 * 4);
  float* mean3 = (float*)take(512 * 4); float* var3 = (float*)take(512 * 4);
  float* embA  = (float*)take((size_t)32 * 512 * 4);
  float* embV  = (float*)take((size_t)128 * 512 * 4);

  // ---- weight re-layout [co][ci][r] -> [co][r][ci], bf16
  auto cw = [&](const float* s, unsigned short* d, int KHW, int n) {
    hipLaunchKernelGGL(cvt_w, dim3((n + 255) / 256), dim3(256), 0, stream, s, d, KHW, n);
  };
  cw(w_tc1, wb1, 3, 786432);
  cw(w_tc2, wb2, 2, 524288);
  cw(w_tc3, wb3, 3, 786432);
  cw(w_spa, wbs, 9, 2359296);

  // ---- activation NCHW f32 -> NHWC bf16
  hipLaunchKernelGGL(cvt_nhwc, dim3(4194304 / 256), dim3(256), 0, stream,
                     audio, abf, 256, 4194304);
  hipLaunchKernelGGL(cvt_nhwc, dim3(12845056 / 256), dim3(256), 0, stream,
                     visual, vbf, 196, 12845056);

  // ---- temp_conv1: 3x1, pad 2, dil 2  -> y1 [32,512,64,4]
  ConvCfg c1 = { abf, wb1, y1, 8192, 1536, 64, 4, 64, 4, 1, 2, 0, 2, 1, 0 };
  launch_conv(c1, stream);
  hipLaunchKernelGGL(bn_stats, dim3(512), dim3(256), 0, stream, y1, 512, 256, 32, mean1, var1);
  hipLaunchKernelGGL(bn_pool_relu, dim3(2097152 / 256), dim3(256), 0, stream,
                     y1, mean1, var1, g1, b1, x2, (float*)nullptr, 512, 64, 4, 2, 2097152);

  // ---- temp_conv2: 1x2, stride (1,2) -> y2 [32,512,32,2]
  ConvCfg c2 = { x2, wb2, y2, 2048, 1024, 32, 2, 32, 4, 2, 0, 0, 1, 2, 0 };
  launch_conv(c2, stream);
  hipLaunchKernelGGL(bn_stats, dim3(512), dim3(256), 0, stream, y2, 512, 64, 32, mean2, var2);
  hipLaunchKernelGGL(bn_pool_relu, dim3(1048576 / 256), dim3(256), 0, stream,
                     y2, mean2, var2, g2, b2, x3, (float*)nullptr, 512, 32, 2, 1, 1048576);

  // ---- temp_conv3: 3x1, pad 1 -> y3 [32,512,32,2]
  ConvCfg c3 = { x3, wb3, y3, 2048, 1536, 32, 2, 32, 2, 1, 1, 0, 1, 1, 0 };
  launch_conv(c3, stream);
  hipLaunchKernelGGL(bn_stats, dim3(512), dim3(256), 0, stream, y3, 512, 64, 32, mean3, var3);
  hipLaunchKernelGGL(bn_pool_relu, dim3(524288 / 256), dim3(256), 0, stream,
                     y3, mean3, var3, g3, b3, (unsigned short*)nullptr, feat_a,
                     512, 32, 2, 2, 524288);

  // ---- spa_conv: 3x3, pad 1, fused ReLU -> feat_v [128,512,14,14]
  ConvCfg cs = { vbf, wbs, feat_v, 25088, 4608, 14, 14, 14, 14, 3, 1, 1, 1, 1, 1 };
  launch_conv(cs, stream);

  // ---- embeddings: global max pools
  hipLaunchKernelGGL(max_reduce, dim3(32 * 512), dim3(64), 0, stream, feat_a, embA, 32);
  hipLaunchKernelGGL(max_reduce, dim3(128 * 512), dim3(64), 0, stream, feat_v, embV, 196);

  // ---- discriminator heads
  hipLaunchKernelGGL(discrim_kernel, dim3(128), dim3(128), 0, stream,
                     embA, embV, w_d1, b_d1, w_d2, b_d2, randidx, perm, common, differ);
}
